// QuantLinearWB_53180285059558
// MI455X (gfx1250) — compile-verified, matrix operand order checked by reading, not run on hardware
//
#include <hip/hip_runtime.h>
#include <hip/hip_bf16.h>

typedef _Float16 v16h __attribute__((ext_vector_type(16)));
typedef _Float16 v4h  __attribute__((ext_vector_type(4)));
typedef float    v8f  __attribute__((ext_vector_type(8)));

#define OUT_F     11008
#define IN_F      4096
#define IN_GROUPS 32
#define PACKED    64
#define BM        128
#define BN        128
#define BK        128              // one quant group per k-tile
#define LDSS      144              // LDS row stride in halves (288B, 32B-multiple)
#define TILE_H    (BM * LDSS)      // halves per tile buffer

__global__ __launch_bounds__(256)
void QuantLinearWB_wmma_kernel(const float* __restrict__ x,
                               const int*   __restrict__ qcodes,
                               const float* __restrict__ scales,
                               const float* __restrict__ zeros,
                               const float* __restrict__ bias,
                               float*       __restrict__ out) {
    // Double-buffered tiles: 4 * 36 KB = 144 KB (CDNA5 WGP has 320 KB LDS)
    alignas(32) __shared__ _Float16 lx[2 * TILE_H];   // x tile, f16
    alignas(32) __shared__ _Float16 lw[2 * TILE_H];   // dequantized W tile, f16

    const int tid  = threadIdx.x;
    const int lane = tid & 31;
    const int wave = tid >> 5;     // 0..7
    const int wm   = wave & 1;     // 2 M-slabs of 64
    const int wn   = wave >> 1;    // 4 N-slabs of 32

    const int bn = blockIdx.x * BN;   // N tile base (out features)
    const int bm = blockIdx.y * BM;   // M tile base (batch*seq)

    v8f acc[4][2];
    {
        v8f z = {};
        #pragma unroll
        for (int im = 0; im < 4; ++im)
            #pragma unroll
            for (int in = 0; in < 2; ++in)
                acc[im][in] = z;
    }

    // ---- per-thread load assignments (constant across groups) ----
    // x: column group fixed per thread, row strides by 8
    const int xc4 = (lane << 2) | ((wave & 0) );   // (tid&31)*4
    const int xc  = (tid & 31) << 2;
    const int xr0 = tid >> 5;                      // + 8*i, i=0..15
    // W: 2 threads per output row, 32 packed words each
    const int wrow  = tid >> 1;                    // 0..127
    const int wbase = (tid & 1) * 32;              // word offset within group
    const int orow  = bn + wrow;
    const int*   qbase = qcodes + ((size_t)orow * IN_GROUPS) * PACKED + wbase;
    const float* srow  = scales + (size_t)orow * IN_GROUPS;
    const float* zrow  = zeros  + (size_t)orow * IN_GROUPS;

    // ---- staging registers for software pipeline ----
    v4h   xreg[16];
    int4  qreg[8];
    float sc = 0.f, zp = 0.f;

    auto load_group = [&](int g) {
        const float* xg = x + (size_t)bm * IN_F + g * BK;
        #pragma unroll
        for (int i = 0; i < 16; ++i) {
            float4 v = *(const float4*)(xg + (size_t)(xr0 + 8 * i) * IN_F + xc);
            xreg[i] = (v4h){(_Float16)v.x, (_Float16)v.y,
                            (_Float16)v.z, (_Float16)v.w};
        }
        const int* qrow = qbase + g * PACKED;
        #pragma unroll
        for (int j = 0; j < 8; ++j)
            qreg[j] = *(const int4*)(qrow + j * 4);
        sc = srow[g];
        zp = zrow[g];
    };

    auto store_group = [&](int p) {
        _Float16* LX = lx + p * TILE_H;
        _Float16* LW = lw + p * TILE_H;
        #pragma unroll
        for (int i = 0; i < 16; ++i)
            *(v4h*)&LX[(xr0 + 8 * i) * LDSS + xc] = xreg[i];
        _Float16* d = &LW[wrow * LDSS + wbase * 2];
        #pragma unroll
        for (int j = 0; j < 8; ++j) {
            int qq[4] = {qreg[j].x, qreg[j].y, qreg[j].z, qreg[j].w};
            #pragma unroll
            for (int t = 0; t < 4; ++t) {
                int hi = (qq[t] >> 4) & 15;   // even k
                int lo =  qq[t]       & 15;   // odd k
                d[j * 8 + t * 2 + 0] = (_Float16)(((float)hi - zp) * sc);
                d[j * 8 + t * 2 + 1] = (_Float16)(((float)lo - zp) * sc);
            }
        }
    };

    const int mrow  = wm * 64 + (lane & 15);
    const int nrow  = wn * 32 + (lane & 15);
    const int khalf = (lane >> 4) * 16;

    auto compute = [&](int p) {
        const _Float16* LX = lx + p * TILE_H;
        const _Float16* LW = lw + p * TILE_H;
        #pragma unroll
        for (int kk = 0; kk < BK; kk += 32) {
            v16h a[4], b[2];
            #pragma unroll
            for (int im = 0; im < 4; ++im)
                a[im] = *(const v16h*)&LX[(mrow + im * 16) * LDSS + kk + khalf];
            #pragma unroll
            for (int in = 0; in < 2; ++in)
                b[in] = *(const v16h*)&LW[(nrow + in * 16) * LDSS + kk + khalf];
            #pragma unroll
            for (int im = 0; im < 4; ++im)
                #pragma unroll
                for (int in = 0; in < 2; ++in)
                    acc[im][in] = __builtin_amdgcn_wmma_f32_16x16x32_f16(
                        false, a[im], false, b[in],
                        (short)0, acc[im][in], false, false);
        }
    };

    // ---- software pipeline: load g+1 before computing g ----
    load_group(0);
    store_group(0);
    __syncthreads();
    for (int g = 0; g < IN_GROUPS; ++g) {
        const int p = g & 1;
        if (g + 1 < IN_GROUPS) load_group(g + 1);   // VMEM under WMMA
        compute(p);
        if (g + 1 < IN_GROUPS) store_group(p ^ 1);  // fill other buffer
        __syncthreads();
    }

    // ---- epilogue: D layout -> global (non-temporal), add bias ----
    // C/D 16x16 f32: VGPR r, lanes 0-15 -> M=r, N=lane; lanes 16-31 -> M=r+8
    const int mbase = bm + wm * 64 + ((lane >> 4) << 3);
    const int nbase = bn + wn * 32 + (lane & 15);
    #pragma unroll
    for (int in = 0; in < 2; ++in) {
        const int   col = nbase + in * 16;
        const float bv  = bias[col];
        #pragma unroll
        for (int im = 0; im < 4; ++im) {
            #pragma unroll
            for (int r = 0; r < 8; ++r) {
                int row = mbase + im * 16 + r;
                // write-once output: NT store keeps x/qcodes resident in L2
                __builtin_nontemporal_store(acc[im][in][r] + bv,
                                            &out[(size_t)row * OUT_F + col]);
            }
        }
    }
}

extern "C" void kernel_launch(void* const* d_in, const int* in_sizes, int n_in,
                              void* d_out, int out_size, void* d_ws, size_t ws_size,
                              hipStream_t stream) {
    const float* x      = (const float*)d_in[0];
    const int*   qcodes = (const int*)  d_in[1];
    const float* scales = (const float*)d_in[2];
    const float* zeros  = (const float*)d_in[3];
    const float* bias   = (const float*)d_in[4];
    float*       out    = (float*)d_out;

    dim3 grid(OUT_F / BN, (4 * 2048) / BM, 1);   // (86, 64)
    dim3 block(256, 1, 1);
    QuantLinearWB_wmma_kernel<<<grid, block, 0, stream>>>(
        x, qcodes, scales, zeros, bias, out);
}
